// SA_Head_24799141167533
// MI455X (gfx1250) — matrix operand order repaired and checked by reading, not compiled, not run on hardware
//
#include <hip/hip_runtime.h>

typedef _Float16 h16;
typedef __attribute__((ext_vector_type(16))) _Float16 v16h;
typedef __attribute__((ext_vector_type(8)))  _Float16 v8h;
typedef __attribute__((ext_vector_type(8)))  float    v8f;

#define N_EMB 256
#define HEAD  64
#define BATCH 4
#define SEQ   4096

static __device__ __forceinline__ v8f wmma_f16(v16h a, v16h b, v8f c) {
  // D = A(16x32 f16) * B(32x16 f16) + C(16x16 f32)
  return __builtin_amdgcn_wmma_f32_16x16x32_f16(
      /*neg_a=*/false, a, /*neg_b=*/false, b,
      /*c_mod=*/(short)0, c, /*reuse_a=*/false, /*reuse_b=*/false);
}

static __device__ __forceinline__ v16h cat8(v8h lo, v8h hi) {
  return __builtin_shufflevector(lo, hi, 0, 1, 2, 3, 4, 5, 6, 7,
                                 8, 9, 10, 11, 12, 13, 14, 15);
}

// ---------------------------------------------------------------------------
// Kernel 0: transpose + f32->f16 convert the three weight matrices.
// W[k=256][n=64] f32  ->  Wt[n=64][k=256] f16, so the projection kernel's
// B-operand (k-major per lane) is a single contiguous 32B load per lane.
// ---------------------------------------------------------------------------
__global__ __launch_bounds__(256)
void sa_wprep_kernel(const float* __restrict__ w0, const float* __restrict__ w1,
                     const float* __restrict__ w2, h16* __restrict__ wt0,
                     h16* __restrict__ wt1, h16* __restrict__ wt2) {
  const int i = blockIdx.x * 256 + threadIdx.x;     // 0 .. 16383
  const int h = i >> 8;                             // / 256
  const int k = i & 255;
  const float* src = (blockIdx.y == 0) ? w0 : (blockIdx.y == 1) ? w1 : w2;
  h16* dst = (blockIdx.y == 0) ? wt0 : (blockIdx.y == 1) ? wt1 : wt2;
  dst[h * N_EMB + k] = (h16)src[k * HEAD + h];
}

// ---------------------------------------------------------------------------
// Kernel 1: Q/K/V projections.  One wave (32 lanes) per 16-row tile of the
// flattened [B*S, 256] embedding.  A-operand (emb rows, fp32->f16) loaded
// once per k-step, reused across 12 WMMAs (3 weights x 4 n-tiles); each
// B-operand is one contiguous v16h load from the transposed f16 weights.
// Outputs: Qh,Kh f16 [B*S][64] row-major; V transposed Vt[B][64][S].
// ---------------------------------------------------------------------------
__global__ __launch_bounds__(32)
void sa_proj_kernel(const float* __restrict__ emb,
                    const h16* __restrict__ wtq,
                    const h16* __restrict__ wtk,
                    const h16* __restrict__ wtv,
                    h16* __restrict__ Qh, h16* __restrict__ Kh,
                    h16* __restrict__ Vth) {
  const int t    = blockIdx.x;          // tile index: rows t*16 .. t*16+15
  const int lane = threadIdx.x;
  const int m    = lane & 15;           // A-row / D-column lane id
  const int g    = lane >> 4;           // lane half

  const int    row0 = t * 16;
  const float* arow = emb + (size_t)(row0 + m) * N_EMB;

  v8f zero = {};
  v8f accq[4], acck[4], accv[4];
#pragma unroll
  for (int i = 0; i < 4; ++i) { accq[i] = zero; acck[i] = zero; accv[i] = zero; }

  for (int ks = 0; ks < 8; ++ks) {      // k over 256 in steps of 32
    const int k0 = ks * 32;
    // A operand: lane half g holds K-chunks {16g..16g+7} and {16g+8..16g+15}
    v16h a;
    {
      const float* plo = arow + k0 + 16 * g;
      const float* phi = arow + k0 + 16 * g + 8;
      v8h lo, hi;
#pragma unroll
      for (int j = 0; j < 8; ++j) { lo[j] = (h16)plo[j]; hi[j] = (h16)phi[j]; }
      a = cat8(lo, hi);
    }
#pragma unroll
    for (int nt = 0; nt < 4; ++nt) {
      const int col = nt * 16 + m;      // output head-dim column for this lane
      const size_t boff = (size_t)col * N_EMB + k0 + g * 16;
      const v16h bq = *(const v16h*)(wtq + boff);
      const v16h bk = *(const v16h*)(wtk + boff);
      const v16h bv = *(const v16h*)(wtv + boff);
      accq[nt] = wmma_f16(a, bq, accq[nt]);
      acck[nt] = wmma_f16(a, bk, acck[nt]);
      accv[nt] = wmma_f16(a, bv, accv[nt]);
    }
  }

  // D layout: lane -> N = m (head col), VGPR r -> M = r + 8g (seq row in tile)
  const int b  = row0 >> 12;            // row0 / 4096
  const int s0 = row0 & (SEQ - 1);
#pragma unroll
  for (int nt = 0; nt < 4; ++nt) {
    const int h = nt * 16 + m;
#pragma unroll
    for (int r = 0; r < 8; ++r) {
      const int srow = row0 + r + 8 * g;            // flat [B*S] row
      Qh[(size_t)srow * HEAD + h] = (h16)accq[nt][r];
      Kh[(size_t)srow * HEAD + h] = (h16)acck[nt][r];
      Vth[((size_t)b * HEAD + h) * SEQ + (s0 + r + 8 * g)] = (h16)accv[nt][r];
    }
  }
}

// ---------------------------------------------------------------------------
// Kernel 2: fused flash attention.  One wave per (batch, 16-query tile).
// Computes S^T = K * Q^T so the score D-tile has N = query: each lane owns
// one query's softmax state; cross-lane traffic is a single shfl_xor(16).
// V-tile loads are issued before the softmax VALU work so they fly under
// independent ALU ops; next block's K/V lines are prefetched explicitly.
// Output GEMM is out^T = V^T * P^T; P^T is rebuilt into B-operand layout
// with 8 shfl_xor(16) ops per 32-key block.
// ---------------------------------------------------------------------------
__global__ __launch_bounds__(32)
void sa_attn_kernel(const h16* __restrict__ Qh, const h16* __restrict__ Kh,
                    const h16* __restrict__ Vth, float* __restrict__ out) {
  const int idx  = blockIdx.x;          // 0..B*(S/16)-1
  const int b    = idx >> 8;            // / (SEQ/16)
  const int q0   = (idx & 255) * 16;
  const int lane = threadIdx.x;
  const int n    = lane & 15;
  const int g    = lane >> 4;

  const h16* Qb = Qh + (size_t)b * SEQ * HEAD;
  const h16* Kb = Kh + (size_t)b * SEQ * HEAD;
  const h16* Vb = Vth + (size_t)b * HEAD * SEQ;

  // Q B-operands (k = head dim, n = query): contiguous 16 f16 per lane.
  v16h qop[2];
  {
    const h16* qrow = Qb + (size_t)(q0 + n) * HEAD;
#pragma unroll
    for (int s = 0; s < 2; ++s)
      qop[s] = *(const v16h*)(qrow + s * 32 + g * 16);
  }

  v8f zero = {};
  v8f acc[4];
#pragma unroll
  for (int i = 0; i < 4; ++i) acc[i] = zero;
  float mi = -1e30f, li = 0.f;
  const float scale = 0.0625f;          // N_EMB**-0.5

  for (int kb = 0; kb < SEQ / 32; ++kb) {
    const int kbase = kb * 32;

    // ---- scores^T for two 16-key subtiles -------------------------------
    v8f st[2];
#pragma unroll
    for (int t = 0; t < 2; ++t) {
      const h16* krow = Kb + (size_t)(kbase + t * 16 + n) * HEAD;
      v8f c = zero;
#pragma unroll
      for (int s = 0; s < 2; ++s) {     // head dim in two 32-chunks
        const v8h lo = *(const v8h*)(krow + s * 32 + 16 * g);
        const v8h hi = *(const v8h*)(krow + s * 32 + 16 * g + 8);
        c = wmma_f16(cat8(lo, hi), qop[s], c);
      }
      st[t] = c;
    }

    // ---- issue V-tile loads early (consumed after softmax) --------------
    v16h av[4];
#pragma unroll
    for (int ht = 0; ht < 4; ++ht) {
      const h16* vrow = Vb + (size_t)(ht * 16 + n) * SEQ + kbase;
      const v8h lo = *(const v8h*)(vrow + 16 * g);
      const v8h hi = *(const v8h*)(vrow + 16 * g + 8);
      av[ht] = cat8(lo, hi);
    }

    // ---- prefetch next 32-key block (speculative, no LOADcnt) -----------
    {
      const int nb = kbase + 32;
      __builtin_prefetch(Kb + (size_t)(nb + lane) * HEAD, 0, 3);       // 32 K rows
      __builtin_prefetch(Vb + (size_t)lane * SEQ + nb, 0, 3);          // V h 0..31
      __builtin_prefetch(Vb + (size_t)(32 + lane) * SEQ + nb, 0, 3);   // V h 32..63
    }

    // ---- online softmax (lane owns query n; halves merged via shfl) -----
    float s0v[8], s1v[8], tmax = -1e30f;
#pragma unroll
    for (int r = 0; r < 8; ++r) {
      s0v[r] = st[0][r] * scale;
      s1v[r] = st[1][r] * scale;
      tmax = fmaxf(tmax, fmaxf(s0v[r], s1v[r]));
    }
    tmax = fmaxf(tmax, __shfl_xor(tmax, 16));
    const float mnew = fmaxf(mi, tmax);
    const float corr = __expf(mi - mnew);
    float p0[8], p1[8], tsum = 0.f;
#pragma unroll
    for (int r = 0; r < 8; ++r) {
      p0[r] = __expf(s0v[r] - mnew);
      p1[r] = __expf(s1v[r] - mnew);
      tsum += p0[r] + p1[r];
    }
    tsum += __shfl_xor(tsum, 16);
    li = li * corr + tsum;
    mi = mnew;
#pragma unroll
    for (int i = 0; i < 4; ++i) acc[i] = acc[i] * corr;

    // ---- reshape P^T (D layout) into 32x16 B operand --------------------
    v16h pb;
#pragma unroll
    for (int j = 0; j < 8; ++j) {
      const float x = (g == 0) ? p1[j] : p0[j];
      const float y = __shfl_xor(x, 16);
      pb[j]     = (h16)((g == 0) ? p0[j] : y);
      pb[8 + j] = (h16)((g == 0) ? y : p1[j]);
    }

    // ---- out^T += V^T-tile * P^T  (4 head tiles, k = 32 keys) -----------
#pragma unroll
    for (int ht = 0; ht < 4; ++ht)
      acc[ht] = wmma_f16(av[ht], pb, acc[ht]);
  }

  // ---- finalize: divide by row sum; lane's 8 h-values are contiguous ----
  const float inv = 1.0f / li;
  float* orow = out + ((size_t)(b * SEQ + q0 + n)) * HEAD;
#pragma unroll
  for (int ht = 0; ht < 4; ++ht)
#pragma unroll
    for (int r = 0; r < 8; ++r)
      orow[ht * 16 + r + 8 * g] = acc[ht][r] * inv;
}

// ---------------------------------------------------------------------------
extern "C" void kernel_launch(void* const* d_in, const int* in_sizes, int n_in,
                              void* d_out, int out_size, void* d_ws, size_t ws_size,
                              hipStream_t stream) {
  // setup_inputs order: ini_emb, wk, wq, wv
  const float* emb = (const float*)d_in[0];
  const float* wk  = (const float*)d_in[1];
  const float* wq  = (const float*)d_in[2];
  const float* wv  = (const float*)d_in[3];
  float* out = (float*)d_out;

  const size_t elems = (size_t)BATCH * SEQ * HEAD;   // 1M f16 each
  const size_t wlen  = (size_t)N_EMB * HEAD;         // 16K f16 each
  h16* Qh  = (h16*)d_ws;
  h16* Kh  = Qh + elems;
  h16* Vth = Kh + elems;
  h16* Wtq = Vth + elems;
  h16* Wtk = Wtq + wlen;
  h16* Wtv = Wtk + wlen;

  dim3 wgrid(wlen / 256, 3);
  sa_wprep_kernel<<<wgrid, 256, 0, stream>>>(wq, wk, wv, Wtq, Wtk, Wtv);

  const int mtiles = BATCH * SEQ / 16;               // 1024
  sa_proj_kernel<<<mtiles, 32, 0, stream>>>(emb, Wtq, Wtk, Wtv, Qh, Kh, Vth);

  const int qtiles = BATCH * (SEQ / 16);             // 1024
  sa_attn_kernel<<<qtiles, 32, 0, stream>>>(Qh, Kh, Vth, out);
}
